// RingConv2d_24446953849120
// MI455X (gfx1250) — matrix operand order repaired and checked by reading, not compiled, not run on hardware
//
#include <hip/hip_runtime.h>
#include <cmath>

// RingConv2d on MI455X (gfx1250): implicit-GEMM 3x3 conv via v_wmma_f32_16x16x32_f16.
//   M = 16 out-channels (A = trig(weights), staged in LDS)
//   N = 16 consecutive output pixels along W (B = cos/sin(x) built per tap)
//   K = 32 "inp" channels ([cos x ; sin x]); 9 taps -> 9 K-steps of 32.
// Each wave computes a 64x16 (channels x pixels) f32 tile (4 accumulators),
// then fuses atan2(dir_y, dir_x) and stores 32x16 outputs.
//
// All trig is expressed as cos(x + phase) so lanes never diverge:
//   sin(x) = cos(x - pi/2), -sin(x) = cos(x + pi/2).

typedef __attribute__((ext_vector_type(16))) _Float16 v16h;
typedef __attribute__((ext_vector_type(8)))  float    v8f;

#define HH 512
#define WW 512
#define CIN 16
#define HWPLANE (HH * WW)

#define PIO2 1.5707963267948966f

__global__ __launch_bounds__(256) void ringconv_wmma_kernel(
    const float* __restrict__ x,      // (8,16,512,512)
    const float* __restrict__ wraw,   // (1,16,32,1,1,3,3) -> flat (16,32,3,3)
    float* __restrict__ out)          // (8,32,512,512)
{
    // ---- Stage trig-transformed weight matrix into LDS: [tap][o(64)][k(32)] f16 ----
    __shared__ _Float16 lds_w[9][64][32];   // 36864 bytes

    for (int e = threadIdx.x; e < 9 * 64 * 32; e += 256) {
        int tap  = e >> 11;           // / 2048
        int rest = e & 2047;
        int o    = rest >> 5;         // 0..63 (kern out-channel)
        int k    = rest & 31;         // 0..31 (inp channel)
        int kh   = tap / 3, kw = tap % 3;
        int i    = k & 15;            // raw input channel
        int oc   = o & 31;            // raw out channel
        // raw weight layout (16,32,3,3): w[o', i] = weight[i][o']
        float wv = wraw[((i * 32 + oc) * 3 + kh) * 3 + kw];
        // kern = [[cos w ; sin w] ; [-sin w ; cos w]] expressed as cos(w + phase):
        //   o<32,k<16: cos(w)      -> phase 0
        //   o<32,k>=16: sin(w)     -> phase -pi/2
        //   o>=32,k<16: -sin(w)    -> phase +pi/2
        //   o>=32,k>=16: cos(w)    -> phase 0
        float phase = (k < 16) ? ((o < 32) ? 0.0f :  PIO2)
                               : ((o < 32) ? -PIO2 : 0.0f);
        lds_w[tap][o][k] = (_Float16)__cosf(wv + phase);
    }
    __syncthreads();

    // ---- Per-wave output tile ----
    const int lane   = threadIdx.x & 31;
    const int wave   = threadIdx.x >> 5;
    const int tile   = blockIdx.x * 8 + wave;        // 131072 tiles total
    const int w0     = (tile & 31) << 4;             // 32 pixel-tiles per row
    const int t2     = tile >> 5;
    const int h      = t2 & 511;
    const int n      = t2 >> 9;
    const int px     = w0 + (lane & 15);             // this lane's output column (N axis)
    const bool hiH   = (lane >= 16);
    // lanes 0-15 produce cos(x) (K=0..15), lanes 16-31 produce sin(x) (K=16..31)
    const float bphase = hiH ? -PIO2 : 0.0f;

    const v8f vzero = {0.f, 0.f, 0.f, 0.f, 0.f, 0.f, 0.f, 0.f};
    v8f acc[4];
    acc[0] = vzero; acc[1] = vzero; acc[2] = vzero; acc[3] = vzero;

    union AView { uint4 q[2]; v16h v; };

    const float* xbase = x + (size_t)n * CIN * HWPLANE;

#pragma unroll
    for (int tap = 0; tap < 9; ++tap) {
        const int kh = tap / 3 - 1;
        const int kw = tap % 3 - 1;
        const int hy = h + kh;
        const int wx = px + kw;
        const bool valid = (hy >= 0) & (hy < HH) & (wx >= 0) & (wx < WW);
        const int hc = min(max(hy, 0), HH - 1);
        const int wc = min(max(wx, 0), WW - 1);
        const float* xp = xbase + (size_t)hc * WW + wc;   // channel 0 of this pixel

        // B operand: 32x16 activations. Zero-pad -> cos(0)=1 / sin(0)=0.
        v16h b;
#pragma unroll
        for (int j = 0; j < 16; ++j) {
            float xv = xp[(size_t)j * HWPLANE];
            xv = valid ? xv : 0.0f;
            b[j] = (_Float16)__cosf(xv + bphase);
        }

        // 4 channel tiles share this B
#pragma unroll
        for (int t = 0; t < 4; ++t) {
            const int row = (t << 4) + (lane & 15);   // out-channel row in [0,64)
            const uint4* rq = (const uint4*)(&lds_w[tap][row][0]);
            AView u;
            // A k order used here (same as B): lane<16 -> K {0..15}; lane>=16 -> K {16..31}
            u.q[0] = rq[hiH ? 2 : 0];
            u.q[1] = rq[hiH ? 3 : 1];
            acc[t] = __builtin_amdgcn_wmma_f32_16x16x32_f16(
                         /*neg_a=*/false, u.v, /*neg_b=*/false, b,
                         /*c_mod=*/(short)0, acc[t],
                         /*reuse_a=*/false, /*reuse_b=*/false);
        }
    }

    // ---- Fused atan2 + store ----
    // C/D layout: VGPR r, lanes 0-15 -> M=r; lanes 16-31 -> M=r+8. M = channel-in-tile.
    // Tiles 0,1 = dir_x (ch 0..31); tiles 2,3 = dir_y (ch 0..31).
    float* obase = out + (size_t)n * 32 * HWPLANE + (size_t)h * WW + px;
#pragma unroll
    for (int t = 0; t < 2; ++t) {
#pragma unroll
        for (int r = 0; r < 8; ++r) {
            float dx = acc[t][r];
            float dy = acc[t + 2][r];
            float v  = atan2f(dy, dx);
            int ch   = (t << 4) + r + (hiH ? 8 : 0);
            obase[(size_t)ch * HWPLANE] = v;
        }
    }
}

extern "C" void kernel_launch(void* const* d_in, const int* in_sizes, int n_in,
                              void* d_out, int out_size, void* d_ws, size_t ws_size,
                              hipStream_t stream) {
    const float* x    = (const float*)d_in[0];
    const float* wgt  = (const float*)d_in[1];
    float*       outp = (float*)d_out;

    // 8 batch * 512 rows * 32 pixel-tiles = 131072 wave-tiles; 8 waves/block
    dim3 grid(131072 / 8);
    dim3 block(256);
    hipLaunchKernelGGL(ringconv_wmma_kernel, grid, block, 0, stream, x, wgt, outp);
}
